// FNO2d_54400055771324
// MI455X (gfx1250) — compile-verified, matrix operand order checked by reading, not run on hardware
//
#include <hip/hip_runtime.h>
#include <hip/hip_bf16.h>
#include <math.h>

typedef __attribute__((ext_vector_type(16))) _Float16 v16h;
typedef __attribute__((ext_vector_type(8)))  _Float16 v8h;
typedef __attribute__((ext_vector_type(8)))  float    v8f;
typedef __attribute__((ext_vector_type(4)))  float    v4f;

__device__ __forceinline__ v8f wmma16(v16h a, v16h b, v8f c) {
  // D = A(16x32 f16) * B(32x16 f16) + C(16x16 f32)
  return __builtin_amdgcn_wmma_f32_16x16x32_f16(false, a, false, b, (short)0, c, false, false);
}
__device__ __forceinline__ v8f vzero8() {
  v8f z = {0.f,0.f,0.f,0.f,0.f,0.f,0.f,0.f};
  return z;
}
__device__ __forceinline__ float gelu_exact(float v) {
  return 0.5f * v * (1.f + erff(v * 0.70710678118654752f));
}

// Fragment loaders. For one lane the fragment's K-offsets are two contiguous
// 8-half runs at +0..7 and +16..23 (relative to its kg base), so with data
// stored K-contiguous per output row these are two aligned b128 loads.
__device__ __forceinline__ v16h fragl(const _Float16* __restrict__ p) {
  v8h a = *(const v8h*)p;          // ds_load_b128
  v8h b = *(const v8h*)(p + 16);   // ds_load_b128
  return __builtin_shufflevector(a, b, 0,1,2,3,4,5,6,7,8,9,10,11,12,13,14,15);
}
__device__ __forceinline__ v16h fragg(const float* __restrict__ p) {
  v4f f0 = *(const v4f*)(p);
  v4f f1 = *(const v4f*)(p + 4);
  v4f f2 = *(const v4f*)(p + 16);
  v4f f3 = *(const v4f*)(p + 20);
  v16h r;
#pragma unroll
  for (int j = 0; j < 4; ++j) {
    r[j]      = (_Float16)f0[j];
    r[4 + j]  = (_Float16)f1[j];
    r[8 + j]  = (_Float16)f2[j];
    r[12 + j] = (_Float16)f3[j];
  }
  return r;
}
__device__ __forceinline__ v16h vneg(v16h x) { return -x; }

__device__ __forceinline__ void stage_copy(const float* __restrict__ src,
                                           _Float16* __restrict__ dst, int n) {
  for (int e = threadIdx.x; e < n; e += 256) dst[e] = (_Float16)src[e];
}

// Generic WMMA GEMM tile: one wave computes a 16-row x (MT*16)-col tile of
//   OUT[row, m] = sum_k IN[row, k] * COEF[k, m]      (optionally complex)
// IN is f32 planar (inR / inI). COEF is f16 in LDS, stored TRANSPOSED as
// [n = output col][k], so B fragments are contiguous b128 LDS loads.
template<int KDIM, int MT, bool CIN, bool COUT>
__device__ __forceinline__ void gemm_tile(const float* __restrict__ inR,
                                          const float* __restrict__ inI,
                                          int inStride, int rowBase,
                                          const _Float16* __restrict__ lcR,
                                          const _Float16* __restrict__ lcI,
                                          v8f* accR, v8f* accI) {
  const int lane = threadIdx.x & 31;
  const int m    = lane & 15;
  const int kg   = (lane >> 4) * 8;
  const float* rR = inR + (long)(rowBase + m) * inStride;
  const float* rI = CIN ? (inI + (long)(rowBase + m) * inStride) : rR;

#pragma unroll
  for (int ks = 0; ks < KDIM; ks += 32) {
    v16h aR = fragg(rR + ks + kg);
    v16h aI, aIn;
    if constexpr (CIN) {
      aI  = fragg(rI + ks + kg);
      aIn = vneg(aI);                 // f16 WMMA has no A-neg modifier
    }
#pragma unroll
    for (int t = 0; t < MT; ++t) {
      int n = t * 16 + m;
      v16h bR = fragl(lcR + n * KDIM + ks + kg);
      v16h bI;
      if constexpr (CIN || COUT) bI = fragl(lcI + n * KDIM + ks + kg);
      accR[t] = wmma16(aR, bR, accR[t]);
      if constexpr (CIN) accR[t] = wmma16(aIn, bI, accR[t]);
      if constexpr (COUT) {
        accI[t] = wmma16(aR, bI, accI[t]);
        if constexpr (CIN) accI[t] = wmma16(aI, bR, accI[t]);
      }
    }
  }
}

// ------------------------------------------------ coefficients (pre-transposed)
__global__ __launch_bounds__(256) void k_coef(float* c1r, float* c1i,
                                              float* c2r, float* c2i,
                                              float* c4r, float* c4i, float* c5) {
  const float PI2 = 6.28318530717958647692f;
  int i = blockIdx.x * 256 + threadIdx.x;
  if (i < 2048) {                       // c1: [m 16][w 128]  forward row rfft
    int mm = i >> 7, w = i & 127;
    float th = PI2 * (float)(w * mm) / 128.f;
    c1r[i] = (mm < 12) ?  cosf(th) : 0.f;
    c1i[i] = (mm < 12) ? -sinf(th) : 0.f;
  } else if (i < 6144) {                // c2: [k1 32][h 128] forward col DFT
    int i2 = i - 2048;
    int k1 = i2 >> 7, h = i2 & 127;
    int f = (k1 < 12) ? k1 : (k1 + 104); // 116..127 = -12..-1
    float th = PI2 * (float)(h * f) / 128.f;
    bool ok = (k1 < 24);
    c2r[i2] = ok ?  cosf(th) : 0.f;
    c2i[i2] = ok ? -sinf(th) : 0.f;
  } else if (i < 10240) {               // c4: [h 128][k1 32] inverse col DFT
    int i3 = i - 6144;
    int h = i3 >> 5, k1 = i3 & 31;
    int f = (k1 < 12) ? k1 : (k1 + 104);
    float th = PI2 * (float)(h * f) / 128.f;
    bool ok = (k1 < 24);
    c4r[i3] = ok ? cosf(th) : 0.f;
    c4i[i3] = ok ? sinf(th) : 0.f;
  } else if (i < 14336) {               // c5: [w 128][k 32] inverse row rfft
    int i4 = i - 10240;                 // k 0..11 = Re modes, 16..27 = Im
    int w = i4 >> 5, k = i4 & 31;
    float v = 0.f;
    const float inv = 1.f / 16384.f;    // 1/(H*W)
    if (k < 12) {
      float wk = (k == 0 ? 1.f : 2.f) * inv;
      v = wk * cosf(PI2 * (float)(k * w) / 128.f);
    } else if (k >= 16 && k < 28) {
      int k2 = k - 16;
      if (k2 > 0) v = -2.f * inv * sinf(PI2 * (float)(k2 * w) / 128.f);
    }
    c5[i4] = v;
  }
}

// ---------------------------------------------------------------- lifting conv
__global__ __launch_bounds__(256) void k_lift(const float* __restrict__ x,
                                              const float* __restrict__ pw,
                                              const float* __restrict__ pb,
                                              float* __restrict__ h0) {
  __shared__ float w[576];
  __shared__ float bsh[32];
  for (int e = threadIdx.x; e < 576; e += 256) w[e] = pw[e];
  if (threadIdx.x < 32) bsh[threadIdx.x] = pb[threadIdx.x];
  __syncthreads();
  int gid = blockIdx.x * 256 + threadIdx.x;   // 64*16384
  int b = gid >> 14, p = gid & 16383;
  int y = p >> 7, xw = p & 127;
  float in[2][3][3];
#pragma unroll
  for (int c = 0; c < 2; ++c)
#pragma unroll
    for (int ky = 0; ky < 3; ++ky)
#pragma unroll
      for (int kx = 0; kx < 3; ++kx) {
        int yy = (y + ky - 1) & 127, xx = (xw + kx - 1) & 127;
        in[c][ky][kx] = x[((b * 2 + c) << 14) + (yy << 7) + xx];
      }
  for (int o = 0; o < 32; ++o) {
    float acc = bsh[o];
#pragma unroll
    for (int c = 0; c < 2; ++c)
#pragma unroll
      for (int ky = 0; ky < 3; ++ky)
#pragma unroll
        for (int kx = 0; kx < 3; ++kx)
          acc += w[((o * 2 + c) * 3 + ky) * 3 + kx] * in[c][ky][kx];
    h0[((b * 32 + o) << 14) + p] = acc;
  }
}

// ------------------------------------------------- K1: row rfft (real->12 cplx)
// rows = (b,c,h) 262144, K=128, M=16 (12 valid). Output transposed: S1[bc][m16][h]
__global__ __launch_bounds__(256) void k_rowdft(const float* __restrict__ h,
                                                const float* __restrict__ c1r,
                                                const float* __restrict__ c1i,
                                                float* __restrict__ s1r,
                                                float* __restrict__ s1i) {
  __shared__ alignas(16) _Float16 lR[16 * 128], lI[16 * 128];
  stage_copy(c1r, lR, 16 * 128);
  stage_copy(c1i, lI, 16 * 128);
  __syncthreads();
  int wid = threadIdx.x >> 5, lane = threadIdx.x & 31;
  int rowBase = (blockIdx.x * 8 + wid) * 16;
  v8f accR[1], accI[1];
  accR[0] = vzero8(); accI[0] = vzero8();
  gemm_tile<128, 1, false, true>(h, nullptr, 128, rowBase, lR, lI, accR, accI);
  int bc = rowBase >> 7, h0 = rowBase & 127;
  int n = lane & 15;
#pragma unroll
  for (int r = 0; r < 8; ++r) {
    int rr = r + 8 * (lane >> 4);
    int idx = (bc * 16 + n) * 128 + h0 + rr;
    s1r[idx] = accR[0][r];
    s1i[idx] = accI[0][r];
  }
}

// ---------------------------------------- K2: column DFT to 24 modes (complex)
// rows = (bc,k2) 32768, K=128 cplx, M=32 (24 valid). Out: X2[k1 24][k2 16][b][c]
__global__ __launch_bounds__(256) void k_coldft(const float* __restrict__ s1r,
                                                const float* __restrict__ s1i,
                                                const float* __restrict__ c2r,
                                                const float* __restrict__ c2i,
                                                float* __restrict__ x2r,
                                                float* __restrict__ x2i) {
  __shared__ alignas(16) _Float16 lR[32 * 128], lI[32 * 128];
  stage_copy(c2r, lR, 32 * 128);
  stage_copy(c2i, lI, 32 * 128);
  __syncthreads();
  int wid = threadIdx.x >> 5, lane = threadIdx.x & 31;
  int rowBase = (blockIdx.x * 8 + wid) * 16;
  v8f accR[2], accI[2];
  accR[0] = accR[1] = vzero8();
  accI[0] = accI[1] = vzero8();
  gemm_tile<128, 2, true, true>(s1r, s1i, 128, rowBase, lR, lI, accR, accI);
  int n = lane & 15;
#pragma unroll
  for (int t = 0; t < 2; ++t) {
    int k1 = t * 16 + n;
    if (k1 >= 24) continue;
#pragma unroll
    for (int r = 0; r < 8; ++r) {
      int gRow = rowBase + r + 8 * (lane >> 4);
      int bc = gRow >> 4, k2 = gRow & 15;
      int b = bc >> 5, c = bc & 31;
      int idx = ((k1 * 16 + k2) * 64 + b) * 32 + c;
      x2r[idx] = accR[t][r];
      x2i[idx] = accI[t][r];
    }
  }
}

// -------------------------------- K3: per-mode spectral channel mix (complex)
// block = one (k1,k2) mode; [64 b x 32 c] x [32 x 32] complex GEMM
__global__ __launch_bounds__(256) void k_mix(const float* __restrict__ x2r,
                                             const float* __restrict__ x2i,
                                             const float* __restrict__ w1r,
                                             const float* __restrict__ w1i,
                                             const float* __restrict__ w2r,
                                             const float* __restrict__ w2i,
                                             float* __restrict__ ybuf, int layer) {
  __shared__ alignas(16) _Float16 wR[32 * 32], wI[32 * 32];   // stored [o][i]
  int mode = blockIdx.x;            // 0..287
  int k1 = mode / 12, k2 = mode % 12;
  const float* sR = (k1 < 12) ? w1r : w2r;
  const float* sI = (k1 < 12) ? w1i : w2i;
  int xm = (k1 < 12) ? k1 : (k1 - 12);
  for (int e = threadIdx.x; e < 1024; e += 256) {
    int o = e >> 5, i = e & 31;
    long idx = ((((long)layer * 32 + i) * 32 + o) * 12 + xm) * 12 + k2;
    wR[e] = (_Float16)sR[idx];
    wI[e] = (_Float16)sI[idx];
  }
  __syncthreads();
  int wid = threadIdx.x >> 5, lane = threadIdx.x & 31;
  int btile = wid >> 1, otile = wid & 1;
  int m = lane & 15, kg = (lane >> 4) * 8;
  const float* rR = x2r + ((k1 * 16 + k2) * 64 + btile * 16 + m) * 32;
  const float* rI = x2i + ((k1 * 16 + k2) * 64 + btile * 16 + m) * 32;
  int n = otile * 16 + m;
  v16h aR  = fragg(rR + kg);
  v16h aI  = fragg(rI + kg);
  v16h aIn = vneg(aI);
  v16h bR  = fragl(wR + n * 32 + kg);
  v16h bI  = fragl(wI + n * 32 + kg);
  v8f accR = vzero8(), accI = vzero8();
  accR = wmma16(aR, bR, accR);
  accR = wmma16(aIn, bI, accR);
  accI = wmma16(aR, bI, accI);
  accI = wmma16(aI, bR, accI);
#pragma unroll
  for (int r = 0; r < 8; ++r) {
    int b = btile * 16 + r + 8 * (lane >> 4);
    int idx = ((b * 32 + n) * 16 + k2) * 64;   // rows (b,o,k2), 64 cols [Re|Im]
    ybuf[idx + k1]      = accR[r];
    ybuf[idx + 32 + k1] = accI[r];
  }
}

// -------------------------- K4: inverse column DFT (24 modes -> 128 h, cplx)
// rows = (b,o,k2) 32768, K=32 cplx (24 valid), M=128 cplx. Out: Z[(bo,h)][32]
__global__ __launch_bounds__(256) void k_icoldft(const float* __restrict__ ybuf,
                                                 const float* __restrict__ c4r,
                                                 const float* __restrict__ c4i,
                                                 float* __restrict__ z) {
  __shared__ alignas(16) _Float16 lR[128 * 32], lI[128 * 32];
  stage_copy(c4r, lR, 128 * 32);
  stage_copy(c4i, lI, 128 * 32);
  __syncthreads();
  int wid = threadIdx.x >> 5, lane = threadIdx.x & 31;
  int rowBase = (blockIdx.x * 8 + wid) * 16;
  v8f accR[8], accI[8];
#pragma unroll
  for (int t = 0; t < 8; ++t) { accR[t] = vzero8(); accI[t] = vzero8(); }
  gemm_tile<32, 8, true, true>(ybuf, ybuf + 32, 64, rowBase, lR, lI, accR, accI);
  int n = lane & 15;
#pragma unroll
  for (int t = 0; t < 8; ++t) {
    int hh = t * 16 + n;
#pragma unroll
    for (int r = 0; r < 8; ++r) {
      int gRow = rowBase + r + 8 * (lane >> 4);
      int bo = gRow >> 4, k2 = gRow & 15;
      int idx = (bo * 128 + hh) * 32;
      z[idx + k2]      = accR[t][r];
      z[idx + 16 + k2] = accI[t][r];
    }
  }
}

// ---------------------- K5: inverse row rfft (12 cplx modes -> 128 real w)
// rows = (b,o,h) 262144, K=32 real (packed Re|Im), M=128 real. Out: XS spatial
__global__ __launch_bounds__(256) void k_irowdft(const float* __restrict__ z,
                                                 const float* __restrict__ c5,
                                                 float* __restrict__ xs) {
  __shared__ alignas(16) _Float16 lR[128 * 32];
  stage_copy(c5, lR, 128 * 32);
  __syncthreads();
  int wid = threadIdx.x >> 5, lane = threadIdx.x & 31;
  int rowBase = (blockIdx.x * 8 + wid) * 16;
  v8f accR[8], accI[8];
#pragma unroll
  for (int t = 0; t < 8; ++t) { accR[t] = vzero8(); }
  gemm_tile<32, 8, false, false>(z, nullptr, 32, rowBase, lR, lR, accR, accI);
  int n = lane & 15;
#pragma unroll
  for (int t = 0; t < 8; ++t)
#pragma unroll
    for (int r = 0; r < 8; ++r) {
      int gRow = rowBase + r + 8 * (lane >> 4);
      xs[(long)gRow * 128 + t * 16 + n] = accR[t][r];
    }
}

// -------------- K6: pointwise 1x1 conv (WMMA) + add spectral + bias + GELU
__global__ __launch_bounds__(256) void k_pw(const float* __restrict__ hIn,
                                            const float* __restrict__ xs,
                                            const float* __restrict__ ww,
                                            const float* __restrict__ wb,
                                            float* __restrict__ hOut, int layer) {
  __shared__ alignas(16) _Float16 hT[256 * 32];   // [pixel][channel]
  __shared__ alignas(16) _Float16 wT[32 * 32];    // [o][i]
  __shared__ float xb[32];
  int b = blockIdx.x >> 6;
  int p0 = (blockIdx.x & 63) << 8;
  const float* hb = hIn + (long)b * (32 * 16384);
  for (int c = 0; c < 32; ++c)
    hT[threadIdx.x * 32 + c] = (_Float16)hb[c * 16384 + p0 + threadIdx.x];
  for (int e = threadIdx.x; e < 1024; e += 256)
    wT[e] = (_Float16)ww[layer * 1024 + e];          // already [o][i]
  if (threadIdx.x < 32) xb[threadIdx.x] = wb[layer * 32 + threadIdx.x];
  __syncthreads();
  int wid = threadIdx.x >> 5, lane = threadIdx.x & 31;
  int m = lane & 15, kg = (lane >> 4) * 8;
  for (int pt = wid; pt < 16; pt += 8) {     // 16 pixel tiles of 16
    v16h bf = fragl(hT + (pt * 16 + m) * 32 + kg);
#pragma unroll
    for (int ot = 0; ot < 2; ++ot) {
      v16h af = fragl(wT + (ot * 16 + m) * 32 + kg);
      v8f acc = vzero8();
      acc = wmma16(af, bf, acc);
      int n = lane & 15;
#pragma unroll
      for (int r = 0; r < 8; ++r) {
        int o = ot * 16 + r + 8 * (lane >> 4);
        int p = p0 + pt * 16 + n;
        long oidx = ((long)(b * 32 + o) << 14) + p;
        float v2 = acc[r] + xs[oidx] + xb[o];
        hOut[oidx] = gelu_exact(v2);
      }
    }
  }
}

// ---------------------------------------------------------------- head
__global__ __launch_bounds__(256) void k_mean(const float* __restrict__ h,
                                              float* __restrict__ g) {
  __shared__ float red[256];
  int bc = blockIdx.x;
  float s = 0.f;
  for (int p = threadIdx.x; p < 16384; p += 256) s += h[((long)bc << 14) + p];
  red[threadIdx.x] = s;
  __syncthreads();
  for (int st = 128; st > 0; st >>= 1) {
    if (threadIdx.x < st) red[threadIdx.x] += red[threadIdx.x + st];
    __syncthreads();
  }
  if (threadIdx.x == 0) g[bc] = red[0] * (1.f / 16384.f);
}

__global__ void k_head(const float* __restrict__ g,
                       const float* __restrict__ q1w, const float* __restrict__ q1b,
                       const float* __restrict__ q2w, const float* __restrict__ q2b,
                       float* __restrict__ out) {
  int b = threadIdx.x;
  if (b >= 64) return;
  float gr[32];
#pragma unroll
  for (int c = 0; c < 32; ++c) gr[c] = g[b * 32 + c];
  float o2 = 0.f;
  for (int j = 0; j < 128; ++j) {
    float s = q1b[j];
#pragma unroll
    for (int c = 0; c < 32; ++c) s += gr[c] * q1w[j * 32 + c];
    o2 += gelu_exact(s) * q2w[j];
  }
  o2 += q2b[0];
  out[b] = 1.f / (1.f + expf(-o2));
}

__global__ __launch_bounds__(256) void k_fill0(float* __restrict__ p, int n) {
  int i = blockIdx.x * 256 + threadIdx.x;
  if (i < n) p[i] = 0.f;
}

// ---------------------------------------------------------------- launch
extern "C" void kernel_launch(void* const* d_in, const int* in_sizes, int n_in,
                              void* d_out, int out_size, void* d_ws, size_t ws_size,
                              hipStream_t stream) {
  const float* x    = (const float*)d_in[0];
  const float* p_w  = (const float*)d_in[1];
  const float* p_b  = (const float*)d_in[2];
  const float* sw1r = (const float*)d_in[3];
  const float* sw1i = (const float*)d_in[4];
  const float* sw2r = (const float*)d_in[5];
  const float* sw2i = (const float*)d_in[6];
  const float* ww   = (const float*)d_in[7];
  const float* wb   = (const float*)d_in[8];
  const float* q1w  = (const float*)d_in[9];
  const float* q1b  = (const float*)d_in[10];
  const float* q2w  = (const float*)d_in[11];
  const float* q2b  = (const float*)d_in[12];
  float* out = (float*)d_out;

  float* w = (float*)d_ws;
  size_t off = 0;
  auto alloc = [&](size_t n) { float* p = w + off; off += n; return p; };
  float* hA  = alloc(33554432);           // [64][32][128][128]
  float* hB  = alloc(33554432);
  float* XS  = alloc(33554432);           // spatial spectral output
  float* S1r = alloc(4194304);            // [bc 2048][m 16][h 128]
  float* S1i = alloc(4194304);
  float* X2r = alloc(786432);             // [k1 24][k2 16][b 64][c 32]
  float* X2i = alloc(786432);
  float* Yb  = alloc(2097152);            // [(b,o,k2) 32768][64: Re|Im over k1]
  float* Z   = alloc(8388608);            // [(b,o,h) 262144][32: Re|Im over k2]
  float* c1r = alloc(2048);  float* c1i = alloc(2048);
  float* c2r = alloc(4096);  float* c2i = alloc(4096);
  float* c4r = alloc(4096);  float* c4i = alloc(4096);
  float* c5  = alloc(4096);
  float* g   = alloc(2048);
  (void)ws_size; (void)in_sizes; (void)n_in; (void)out_size;

  k_coef<<<96, 256, 0, stream>>>(c1r, c1i, c2r, c2i, c4r, c4i, c5);
  k_lift<<<4096, 256, 0, stream>>>(x, p_w, p_b, hA);

  float* cur = hA;
  float* nxt = hB;
  for (int l = 0; l < 4; ++l) {
    k_fill0<<<8192, 256, 0, stream>>>(Yb, 2097152);
    k_rowdft<<<2048, 256, 0, stream>>>(cur, c1r, c1i, S1r, S1i);
    k_coldft<<<256, 256, 0, stream>>>(S1r, S1i, c2r, c2i, X2r, X2i);
    k_mix<<<288, 256, 0, stream>>>(X2r, X2i, sw1r, sw1i, sw2r, sw2i, Yb, l);
    k_icoldft<<<256, 256, 0, stream>>>(Yb, c4r, c4i, Z);
    k_irowdft<<<2048, 256, 0, stream>>>(Z, c5, XS);
    k_pw<<<4096, 256, 0, stream>>>(cur, XS, ww, wb, nxt, l);
    float* t = cur; cur = nxt; nxt = t;
  }
  k_mean<<<2048, 256, 0, stream>>>(cur, g);
  k_head<<<1, 64, 0, stream>>>(g, q1w, q1b, q2w, q2b, out);
}
